// VelocityNet_9268539424828
// MI455X (gfx1250) — matrix operand order repaired and checked by reading, not compiled, not run on hardware
//
#include <hip/hip_runtime.h>
#include <hip/hip_bf16.h>

// ---------------------------------------------------------------------------
// VelocityNet for MI455X (gfx1250, wave32, WMMA).
// All GEMMs run on v_wmma_f32_16x16x32_bf16 (fp32 data -> bf16 operands,
// fp32 accum). Edge MLP (E=256000, K=289->320, N=128) dominates FLOPs; h is
// mirrored in bf16 so edge gathers stay L2-resident (5MB << 192MB L2).
// Edge kernel: 32 edges/wave, each B fragment feeds two WMMAs; msg scatters
// go directly from accumulator registers via global_atomic_add_f32.
// h[src]/h[dst] gathers use GLOBAL_LOAD_ASYNC_TO_LDS_B128 (ASYNCcnt).
// ---------------------------------------------------------------------------

#define NN   20000
#define NE   256000
#define IND  64
#define HID  128
#define EED  32
#define NL   3

typedef __attribute__((ext_vector_type(16))) __bf16 v16bf;
typedef __attribute__((ext_vector_type(8)))  __bf16 v8bf;
typedef __attribute__((ext_vector_type(4)))  __bf16 v4bf;
typedef __attribute__((ext_vector_type(8)))  float  v8f;
typedef __attribute__((ext_vector_type(4)))  int    v4i;

#if __has_builtin(__builtin_amdgcn_global_load_async_to_lds_b128)
#define USE_ASYNC_GATHER 1
// Parameter types per clang diagnostic: (v4i __device__*, v4i __shared__*, imm, imm)
typedef __attribute__((address_space(1))) v4i* gbl_b128_p;
typedef __attribute__((address_space(3))) v4i* lds_b128_p;
#else
#define USE_ASYNC_GATHER 0
#endif

// --- WMMA fragment loaders (contiguous k order, same in A and B) -------------------
// A (16x32 bf16): lane m = lane&15, group g = lane>>4 holds 16 contiguous
// elements: K = kt*32 + g*16 .. +15 (matches packed B order).
__device__ __forceinline__ v16bf load_a_frag(const __bf16* A, int kpad, int row,
                                             int kt, int g) {
  const v8bf lo = *(const v8bf*)(A + (size_t)row * kpad + kt * 32 + g * 16);
  const v8bf hi = *(const v8bf*)(A + (size_t)row * kpad + kt * 32 + g * 16 + 8);
  return __builtin_shufflevector(lo, hi, 0,1,2,3,4,5,6,7,8,9,10,11,12,13,14,15);
}
// B (32x16 bf16): prepacked so each lane's 16 values (k = kt*32 + (lane>>4)*16 + j,
// n = nt*16 + (lane&15)) are 32 contiguous bytes.
__device__ __forceinline__ v16bf load_b_frag(const __bf16* Bp, int tileIdx, int lane) {
  return *(const v16bf*)(Bp + ((size_t)tileIdx * 32 + lane) * 16);
}
__device__ __forceinline__ float silu_f(float v) {
  return v / (1.0f + __expf(-v));
}

// --- small utility kernels --------------------------------------------------
__global__ void zero_f32_kernel(float* p, int n) {
  int i = blockIdx.x * blockDim.x + threadIdx.x;
  if (i < n) p[i] = 0.0f;
}

__global__ void copy_x_kernel(const float* __restrict__ X, float* __restrict__ x) {
  int i = blockIdx.x * blockDim.x + threadIdx.x;
  if (i < NN * 3) x[i] = X[i];
}

__global__ void count_kernel(const int* __restrict__ edges, float* __restrict__ cnt) {
  int e = blockIdx.x * blockDim.x + threadIdx.x;
  if (e < NE) atomicAdd(&cnt[edges[NE + e]], 1.0f);
}

// Pack fp32 (Korig x NC, row-major) weight into bf16 WMMA-B fragment order,
// zero-padded to KT*32 x NT*16.
__global__ void pack_w_kernel(const float* __restrict__ W, __bf16* __restrict__ out,
                              int Korig, int NC, int KT, int NT) {
  int id = blockIdx.x * blockDim.x + threadIdx.x;
  int total = KT * NT * 32 * 16;
  if (id >= total) return;
  int j    = id & 15;
  int lane = (id >> 4) & 31;
  int t    = id >> 9;            // kt*NT + nt
  int nt   = t % NT;
  int kt   = t / NT;
  int k = kt * 32 + (lane >> 4) * 16 + j;
  int n = nt * 16 + (lane & 15);
  float v = (k < Korig && n < NC) ? W[(size_t)k * NC + n] : 0.0f;
  out[id] = (__bf16)v;
}

// feat = [H_t(64) | cond(128) | sin(t*f)(64) | cos(t*f)(64)] as bf16, K=320.
__global__ void feat_kernel(const float* __restrict__ Ht, const float* __restrict__ cond,
                            const float* __restrict__ t, __bf16* __restrict__ feat) {
  int id = blockIdx.x * blockDim.x + threadIdx.x;
  if (id >= NN * 320) return;
  int n = id / 320, c = id % 320;
  float v;
  if (c < 64) {
    v = Ht[n * 64 + c];
  } else if (c < 192) {
    v = cond[n * 128 + (c - 64)];
  } else {
    int j = (c < 256) ? (c - 192) : (c - 256);
    float f   = __expf((float)j * (-9.210340371976184f / 63.0f));
    float ang = t[n] * f;
    v = (c < 256) ? __sinf(ang) : __cosf(ang);
  }
  feat[id] = (__bf16)v;
}

// --- generic WMMA GEMM: C = act(A @ Bp + bias), one wave per 16-row M tile --
__global__ void gemm_wmma_kernel(const __bf16* __restrict__ A, const __bf16* __restrict__ Bp,
                                 const float* __restrict__ bias,
                                 float* __restrict__ Cf, __bf16* __restrict__ Cb,
                                 int Mtiles, int kpad, int ktiles, int ntiles, int do_relu) {
  const int lane = threadIdx.x & 31;
  const int wave = threadIdx.x >> 5;
  const int tile = blockIdx.x * (blockDim.x >> 5) + wave;
  if (tile >= Mtiles) return;
  const int g = lane >> 4, nl = lane & 15, mbase = g * 8;
  const int arow = tile * 16 + nl;
  const int NC = ntiles * 16;
  for (int nt = 0; nt < ntiles; ++nt) {
    v8f acc = {};
    for (int kt = 0; kt < ktiles; ++kt) {
      v16bf a = load_a_frag(A, kpad, arow, kt, g);
      v16bf b = load_b_frag(Bp, kt * ntiles + nt, lane);
      acc = __builtin_amdgcn_wmma_f32_16x16x32_bf16(false, a, false, b,
                                                    (short)0, acc, false, false);
    }
    const int col = nt * 16 + nl;
    const float bv = bias ? bias[col] : 0.0f;
#pragma unroll
    for (int r = 0; r < 8; ++r) {
      float v = acc[r] + bv;
      if (do_relu) v = v > 0.0f ? v : 0.0f;
      const int row = tile * 16 + mbase + r;
      if (Cf) Cf[(size_t)row * NC + col] = v;
      if (Cb) Cb[(size_t)row * NC + col] = (__bf16)v;
    }
  }
}

// --- edge message kernel: one wave per 32-edge tile -------------------------
// m_in = [h[src] | h[dst] | d2 | eemb | 0pad] (K=320); msg = silu(m_in@We+be);
// w = msg@Wx+bx. Two 16-edge sub-tiles share every B fragment (2 WMMAs per
// weight load). msg scatters to agg_m straight from C-fragment registers.
#define KPE 320
__global__ void __launch_bounds__(64)
edge_kernel(const __bf16* __restrict__ hb, const float* __restrict__ x,
            const int* __restrict__ edges, const int* __restrict__ etype,
            const float* __restrict__ etable,
            const __bf16* __restrict__ Wep, const float* __restrict__ be,
            const float* __restrict__ Wx, const float* __restrict__ bx,
            float* __restrict__ agg_x, float* __restrict__ agg_m) {
  __shared__ __bf16 sA[2][32 * KPE];     // 40.0 KB
  __shared__ float  sDiff[2][32 * 3];
  const int lane = threadIdx.x & 31;
  const int w    = threadIdx.x >> 5;
  const int e0   = (blockIdx.x * 2 + w) * 32;
  const int e    = e0 + lane;
  const int sIdx = edges[e];
  const int dIdx = edges[NE + e];

  // gather this lane's edge row: [h[src] | h[dst]] (16B chunks; L2-hot)
  {
    const __bf16* srcp = hb + (size_t)sIdx * HID;
    const __bf16* dstp = hb + (size_t)dIdx * HID;
    __bf16* row = &sA[w][lane * KPE];
#if USE_ASYNC_GATHER
#pragma unroll
    for (int j = 0; j < 16; ++j) {
      __builtin_amdgcn_global_load_async_to_lds_b128(
          (gbl_b128_p)(srcp + j * 8), (lds_b128_p)(row + j * 8), 0, 0);
      __builtin_amdgcn_global_load_async_to_lds_b128(
          (gbl_b128_p)(dstp + j * 8), (lds_b128_p)(row + HID + j * 8), 0, 0);
    }
#else
    const v8bf* sp = (const v8bf*)srcp;
    const v8bf* dp = (const v8bf*)dstp;
    v8bf* r0 = (v8bf*)row;
    v8bf* r1 = (v8bf*)(row + HID);
#pragma unroll
    for (int j = 0; j < 16; ++j) { r0[j] = sp[j]; r1[j] = dp[j]; }
#endif
  }
  // diff, d2, eemb, zero-pad (one edge per lane)
  {
    float dx = x[sIdx * 3 + 0] - x[dIdx * 3 + 0];
    float dy = x[sIdx * 3 + 1] - x[dIdx * 3 + 1];
    float dz = x[sIdx * 3 + 2] - x[dIdx * 3 + 2];
    sDiff[w][lane * 3 + 0] = dx;
    sDiff[w][lane * 3 + 1] = dy;
    sDiff[w][lane * 3 + 2] = dz;
    __bf16* rowp = &sA[w][lane * KPE];
    rowp[256] = (__bf16)(dx * dx + dy * dy + dz * dz);
    const int et = etype[e];
#pragma unroll
    for (int j = 0; j < EED; ++j) rowp[257 + j] = (__bf16)etable[et * EED + j];
    for (int j = 289; j < KPE; ++j) rowp[j] = (__bf16)0.0f;
  }
#if USE_ASYNC_GATHER
#if __has_builtin(__builtin_amdgcn_s_wait_asynccnt)
  __builtin_amdgcn_s_wait_asynccnt(0);
#else
  asm volatile("s_wait_asynccnt 0" ::: "memory");
#endif
#endif
  __syncthreads();

  const int g = lane >> 4, nl = lane & 15, mbase = g * 8;
  // destination node ids for the C-fragment rows this lane owns
  int dA[8], dB[8];
#pragma unroll
  for (int r = 0; r < 8; ++r) {
    dA[r] = edges[NE + e0 + mbase + r];
    dB[r] = edges[NE + e0 + 16 + mbase + r];
  }
  float waccA[8] = {0, 0, 0, 0, 0, 0, 0, 0};
  float waccB[8] = {0, 0, 0, 0, 0, 0, 0, 0};
  for (int nt = 0; nt < 8; ++nt) {
    v8f accA = {}, accB = {};
    for (int kt = 0; kt < 10; ++kt) {
      v16bf b  = load_b_frag(Wep, kt * 8 + nt, lane);
      v16bf aA = load_a_frag(&sA[w][0], KPE, nl, kt, g);
      v16bf aB = load_a_frag(&sA[w][0], KPE, 16 + nl, kt, g);
      accA = __builtin_amdgcn_wmma_f32_16x16x32_bf16(false, aA, false, b,
                                                     (short)0, accA, false, false);
      accB = __builtin_amdgcn_wmma_f32_16x16x32_bf16(false, aB, false, b,
                                                     (short)0, accB, false, false);
    }
    const int col = nt * 16 + nl;
    const float bv = be[col], wxv = Wx[col];
#pragma unroll
    for (int r = 0; r < 8; ++r) {
      float mA = silu_f(accA[r] + bv);
      float mB = silu_f(accB[r] + bv);
      waccA[r] += mA * wxv;
      waccB[r] += mB * wxv;
      atomicAdd(&agg_m[(size_t)dA[r] * HID + col], mA);
      atomicAdd(&agg_m[(size_t)dB[r] * HID + col], mB);
    }
  }
  // reduce msg@Wx across the 16 lanes of each half-wave (same 8 rows)
#pragma unroll
  for (int off = 1; off < 16; off <<= 1) {
#pragma unroll
    for (int r = 0; r < 8; ++r) {
      waccA[r] += __shfl_xor(waccA[r], off, 32);
      waccB[r] += __shfl_xor(waccB[r], off, 32);
    }
  }
  const float bxv = bx[0];
  if (nl == 0) {  // lanes 0 and 16 own rows mbase..mbase+7 of both sub-tiles
    for (int r = 0; r < 8; ++r) {
      const int rowA = mbase + r, rowB = 16 + mbase + r;
      const float weA = waccA[r] + bxv;
      const float weB = waccB[r] + bxv;
      for (int c = 0; c < 3; ++c) {
        atomicAdd(&agg_x[dA[r] * 3 + c], sDiff[w][rowA * 3 + c] * weA);
        atomicAdd(&agg_x[dB[r] * 3 + c], sDiff[w][rowB * 3 + c] * weB);
      }
    }
  }
}

// --- node update: h += silu([h|agg_m]@Wh+bh); x += agg_x/(cnt+1) ------------
__global__ void __launch_bounds__(128)
node_update_kernel(float* __restrict__ h, __bf16* __restrict__ hb,
                   const float* __restrict__ agg_m, const float* __restrict__ agg_x,
                   const float* __restrict__ cnt, float* __restrict__ x,
                   const __bf16* __restrict__ Whp, const float* __restrict__ bh) {
  __shared__ __bf16 sA[4][16 * 256];  // 32 KB
  const int lane = threadIdx.x & 31;
  const int w    = threadIdx.x >> 5;
  const int tile = blockIdx.x * 4 + w;
  if (tile >= NN / 16) return;
  // build [h | agg_m] rows in bf16 (float4 loads, 8B bf16 stores)
  for (int c4 = lane; c4 < 16 * 64; c4 += 32) {
    const int row = c4 >> 6;
    const int col = (c4 & 63) * 4;
    const int node = tile * 16 + row;
    float4 v = (col < HID)
                   ? *(const float4*)(h + (size_t)node * HID + col)
                   : *(const float4*)(agg_m + (size_t)node * HID + (col - HID));
    v4bf o;
    o[0] = (__bf16)v.x; o[1] = (__bf16)v.y; o[2] = (__bf16)v.z; o[3] = (__bf16)v.w;
    *(v4bf*)(&sA[w][row * 256 + col]) = o;
  }
  __syncthreads();
  const int g = lane >> 4, nl = lane & 15, mbase = g * 8;
  for (int nt = 0; nt < 8; ++nt) {
    v8f acc = {};
    for (int kt = 0; kt < 8; ++kt) {
      v16bf a = load_a_frag(&sA[w][0], 256, nl, kt, g);
      v16bf b = load_b_frag(Whp, kt * 8 + nt, lane);
      acc = __builtin_amdgcn_wmma_f32_16x16x32_bf16(false, a, false, b,
                                                    (short)0, acc, false, false);
    }
    const int col = nt * 16 + nl;
    const float bv = bh[col];
#pragma unroll
    for (int r = 0; r < 8; ++r) {
      const int row = tile * 16 + mbase + r;
      float hv = h[(size_t)row * HID + col] + silu_f(acc[r] + bv);
      h[(size_t)row * HID + col]  = hv;
      hb[(size_t)row * HID + col] = (__bf16)hv;
    }
  }
  if (lane < 16) {
    const int node = tile * 16 + lane;
    const float inv = 1.0f / (cnt[node] + 1.0f);
    for (int c = 0; c < 3; ++c) x[node * 3 + c] += agg_x[node * 3 + c] * inv;
  }
}

// --- output: v_H = mask*(h@h2i+b), v_X = mask*x -----------------------------
__global__ void __launch_bounds__(128)
out_kernel(const __bf16* __restrict__ hb, const float* __restrict__ x,
           const __bf16* __restrict__ h2ip, const float* __restrict__ h2ib,
           const unsigned char* __restrict__ mask, float* __restrict__ out) {
  const int lane = threadIdx.x & 31;
  const int w    = threadIdx.x >> 5;
  const int tile = blockIdx.x * 4 + w;
  if (tile >= NN / 16) return;
  const int g = lane >> 4, nl = lane & 15, mbase = g * 8;
  const int arow = tile * 16 + nl;
  for (int nt = 0; nt < 4; ++nt) {
    v8f acc = {};
    for (int kt = 0; kt < 4; ++kt) {
      v16bf a = load_a_frag(hb, HID, arow, kt, g);
      v16bf b = load_b_frag(h2ip, kt * 4 + nt, lane);
      acc = __builtin_amdgcn_wmma_f32_16x16x32_bf16(false, a, false, b,
                                                    (short)0, acc, false, false);
    }
    const int col = nt * 16 + nl;
    const float bv = h2ib[col];
#pragma unroll
    for (int r = 0; r < 8; ++r) {
      const int row = tile * 16 + mbase + r;
      out[(size_t)row * IND + col] = mask[row] ? (acc[r] + bv) : 0.0f;
    }
  }
  if (lane < 16) {
    const int node = tile * 16 + lane;
    const float m = mask[node] ? 1.0f : 0.0f;
    for (int c = 0; c < 3; ++c)
      out[(size_t)NN * IND + node * 3 + c] = m * x[node * 3 + c];
  }
}

// ---------------------------------------------------------------------------
static inline size_t align256(size_t v) { return (v + 255u) & ~size_t(255u); }

extern "C" void kernel_launch(void* const* d_in, const int* in_sizes, int n_in,
                              void* d_out, int out_size, void* d_ws, size_t ws_size,
                              hipStream_t stream) {
  (void)in_sizes; (void)n_in; (void)out_size; (void)ws_size;
  // inputs (setup_inputs dict order)
  const float* Ht     = (const float*)d_in[0];
  const float* Xt     = (const float*)d_in[1];
  const float* cond   = (const float*)d_in[2];
  const float* tvec   = (const float*)d_in[3];
  const int*   edges  = (const int*)d_in[4];
  const int*   etype  = (const int*)d_in[5];
  const unsigned char* gmask = (const unsigned char*)d_in[6];
  // d_in[7] = batch_ids (unused by reference)
  const float* W0 = (const float*)d_in[8];
  const float* b0 = (const float*)d_in[9];
  const float* W1 = (const float*)d_in[10];
  const float* b1 = (const float*)d_in[11];
  const float* W2 = (const float*)d_in[12];
  const float* b2 = (const float*)d_in[13];
  const float* etable = (const float*)d_in[14];
  const float* We = (const float*)d_in[15];
  const float* be = (const float*)d_in[16];
  const float* Wx = (const float*)d_in[17];
  const float* bx = (const float*)d_in[18];
  const float* Wh = (const float*)d_in[19];
  const float* bh = (const float*)d_in[20];
  const float* h2iW = (const float*)d_in[21];
  const float* h2ib = (const float*)d_in[22];
  float* out = (float*)d_out;

  // workspace carve-out
  char* base = (char*)d_ws;
  size_t o = 0;
  float*  h_f32  = (float*)(base + o);  o = align256(o + (size_t)NN * HID * 4);
  __bf16* h_b16  = (__bf16*)(base + o); o = align256(o + (size_t)NN * HID * 2);
  float*  x_f32  = (float*)(base + o);  o = align256(o + (size_t)NN * 3 * 4);
  float*  cnt    = (float*)(base + o);  o = align256(o + (size_t)NN * 4);
  float*  agg_x  = (float*)(base + o);  o = align256(o + (size_t)NN * 3 * 4);
  float*  agg_m  = (float*)(base + o);  o = align256(o + (size_t)NN * HID * 4);
  __bf16* feat   = (__bf16*)(base + o); o = align256(o + (size_t)NN * 320 * 2);
  __bf16* a0     = (__bf16*)(base + o); o = align256(o + (size_t)NN * HID * 2);
  __bf16* a1     = (__bf16*)(base + o); o = align256(o + (size_t)NN * HID * 2);
  const int szW0  = 10 * 8 * 512;  // KT*NT*32*16 packed bf16 elements
  const int szW12 = 4 * 8 * 512;
  const int szWe  = 10 * 8 * 512;
  const int szWh  = 8 * 8 * 512;
  const int szH2I = 4 * 4 * 512;
  __bf16* W0p  = (__bf16*)(base + o); o = align256(o + (size_t)szW0 * 2);
  __bf16* W1p  = (__bf16*)(base + o); o = align256(o + (size_t)szW12 * 2);
  __bf16* W2p  = (__bf16*)(base + o); o = align256(o + (size_t)szW12 * 2);
  __bf16* Wep  = (__bf16*)(base + o); o = align256(o + (size_t)szWe * 3 * 2);
  __bf16* Whp  = (__bf16*)(base + o); o = align256(o + (size_t)szWh * 3 * 2);
  __bf16* H2Ip = (__bf16*)(base + o); o = align256(o + (size_t)szH2I * 2);

  const int TPB = 256;
  const int Mtiles = NN / 16;                       // 1250 exact
  const int gemmBlocks = (Mtiles + 3) / 4;          // 4 waves/block

  // init: cnt, aggregates, x
  zero_f32_kernel<<<(NN + TPB - 1) / TPB, TPB, 0, stream>>>(cnt, NN);
  zero_f32_kernel<<<(NN * 3 + TPB - 1) / TPB, TPB, 0, stream>>>(agg_x, NN * 3);
  zero_f32_kernel<<<(NN * HID + TPB - 1) / TPB, TPB, 0, stream>>>(agg_m, NN * HID);
  copy_x_kernel<<<(NN * 3 + TPB - 1) / TPB, TPB, 0, stream>>>(Xt, x_f32);
  count_kernel<<<(NE + TPB - 1) / TPB, TPB, 0, stream>>>(edges, cnt);

  // pack weights into WMMA-B fragment order
  pack_w_kernel<<<(szW0 + TPB - 1) / TPB, TPB, 0, stream>>>(W0, W0p, 320, 128, 10, 8);
  pack_w_kernel<<<(szW12 + TPB - 1) / TPB, TPB, 0, stream>>>(W1, W1p, 128, 128, 4, 8);
  pack_w_kernel<<<(szW12 + TPB - 1) / TPB, TPB, 0, stream>>>(W2, W2p, 128, 128, 4, 8);
  for (int l = 0; l < NL; ++l) {
    pack_w_kernel<<<(szWe + TPB - 1) / TPB, TPB, 0, stream>>>(
        We + (size_t)l * 289 * 128, Wep + (size_t)l * szWe, 289, 128, 10, 8);
    pack_w_kernel<<<(szWh + TPB - 1) / TPB, TPB, 0, stream>>>(
        Wh + (size_t)l * 256 * 128, Whp + (size_t)l * szWh, 256, 128, 8, 8);
  }
  pack_w_kernel<<<(szH2I + TPB - 1) / TPB, TPB, 0, stream>>>(h2iW, H2Ip, 128, 64, 4, 4);

  // node MLP: feat -> a0 -> a1 -> h
  feat_kernel<<<(NN * 320 + TPB - 1) / TPB, TPB, 0, stream>>>(Ht, cond, tvec, feat);
  gemm_wmma_kernel<<<gemmBlocks, 128, 0, stream>>>(feat, W0p, b0, nullptr, a0,
                                                   Mtiles, 320, 10, 8, 1);
  gemm_wmma_kernel<<<gemmBlocks, 128, 0, stream>>>(a0, W1p, b1, nullptr, a1,
                                                   Mtiles, 128, 4, 8, 1);
  gemm_wmma_kernel<<<gemmBlocks, 128, 0, stream>>>(a1, W2p, b2, h_f32, h_b16,
                                                   Mtiles, 128, 4, 8, 0);

  // message-passing layers
  const int edgeBlocks = (NE / 32) / 2;             // 8000 tiles of 32, 2 waves/block
  for (int l = 0; l < NL; ++l) {
    edge_kernel<<<edgeBlocks, 64, 0, stream>>>(
        h_b16, x_f32, edges, etype, etable,
        Wep + (size_t)l * szWe, be + l * HID, Wx + l * HID, bx + l,
        agg_x, agg_m);
    node_update_kernel<<<gemmBlocks, 128, 0, stream>>>(
        h_f32, h_b16, agg_m, agg_x, cnt, x_f32,
        Whp + (size_t)l * szWh, bh + l * HID);
    if (l + 1 < NL) {
      zero_f32_kernel<<<(NN * 3 + TPB - 1) / TPB, TPB, 0, stream>>>(agg_x, NN * 3);
      zero_f32_kernel<<<(NN * HID + TPB - 1) / TPB, TPB, 0, stream>>>(agg_m, NN * HID);
    }
  }

  // outputs: v_H (N x 64) then v_X (N x 3)
  out_kernel<<<gemmBlocks, 128, 0, stream>>>(h_b16, x_f32, H2Ip, h2ib, gmask, out);
}
